// real_space_DNN_conv_71674414235684
// MI455X (gfx1250) — matrix operand order repaired and checked by reading, not compiled, not run on hardware
//
#include <hip/hip_runtime.h>
#include <hip/hip_bf16.h>
#include <math.h>
#include <stdint.h>

// ---------------------------------------------------------------------------
// D2NN forward on MI455X (gfx1250), FFT-as-matmul on WMMA.
// Per propagation (batch 64, one shared transfer function H since
// IMG_DIST == LAYER_DIST):
//   T  = Fc * X^T'        (384x128) K=128   (zero-pad folded into K)
//   Y  = T  * Fc^T        (384x384) K=128
//   Y *= H ; transpose -> Yt (f16)
//   Z  = Gc * Yt^T        (128x384) K=384   (center crop folded into M)
//   Xn = Z  * Gc^T        (128x128) K=384
//   phase mask / snapshot / |.|^2
// Every GEMM is C = A * B^T, f16 operands / f32 accumulate.  Tiles are
// streamed with global_load_async_to_lds_b128 into ping-pong LDS buffers
// (DMA overlapped with WMMA via bounded s_wait_asynccnt), fragments are
// 2x ds_load_b128 per lane, and the complex MAC uses 3-product Karatsuba
// (48 WMMAs per 64x64x32 step instead of 64).
// ---------------------------------------------------------------------------

#define NPX    128
#define LDIM   384
#define BATCH  64
#define NLAY   5
#define PIXSZ  1.0e-4
#define WLEN   5.32e-7
#define DISTZ  0.005
#define TWO_PI 6.283185307179586476925286766559
#define PI_D   3.141592653589793238462643383279

typedef __attribute__((ext_vector_type(16))) _Float16 v16h;
typedef __attribute__((ext_vector_type(8)))  _Float16 v8h;
typedef __attribute__((ext_vector_type(8)))  float    v8f;

// Low 32 bits of a generic shared-memory address = LDS byte address.
__device__ __forceinline__ uint32_t lds_off(const void* p) {
  return (uint32_t)(uintptr_t)p;
}

__device__ __forceinline__ void async_cp16(uint32_t lds, unsigned long long g) {
#if defined(__gfx1250__)
  asm volatile("global_load_async_to_lds_b128 %0, %1, off"
               :: "v"(lds), "v"(g) : "memory");
#endif
}

__device__ __forceinline__ void wait_async(int pending) {
#if defined(__gfx1250__)
  if (pending) asm volatile("s_wait_asynccnt 0x8" ::: "memory");
  else         asm volatile("s_wait_asynccnt 0x0" ::: "memory");
#endif
}

#define WMMA_F16(a, b, c) \
  __builtin_amdgcn_wmma_f32_16x16x32_f16(false, (a), false, (b), (short)0, (c), false, false)

// ---------------------------------------------------------------------------
// Complex GEMM: C = A * B^T (split planes, f16 in / f32 accumulate).
// A:  M x K row-major f16 (lda, halves).   Bt: N x K row-major f16 (ldb).
// Block = 128 threads (4 waves), block tile 64x64, each wave 2x2 WMMA tiles.
// Outputs: f16 planes (C16*) and/or f32 planes (C32*), either may be null.
// ---------------------------------------------------------------------------
#define LROW 40                       // padded LDS row stride (halves)
#define LPLN (64 * LROW)              // LDS plane stride (halves)

__global__ __launch_bounds__(128)
void k_cgemm(const _Float16* __restrict__ Aar, const _Float16* __restrict__ Aai,
             long long sA, int lda,
             const _Float16* __restrict__ Btr, const _Float16* __restrict__ Bti,
             long long sB, int ldb,
             _Float16* __restrict__ C16r, _Float16* __restrict__ C16i,
             float* __restrict__ C32r, float* __restrict__ C32i,
             long long sC, int ldc, int K)
{
  __shared__ alignas(16) _Float16 Ls[2][4 * LPLN];   // ping-pong, 40 KB

  const int tid  = threadIdx.x;
  const int lane = tid & 31;
  const int wv   = tid >> 5;
  const int wm   = wv >> 1, wn = wv & 1;             // wave tile origin /32
  const int bm   = blockIdx.y, bn = blockIdx.x, b = blockIdx.z;

  const _Float16* pA[2] = { Aar + (size_t)b * sA, Aai + (size_t)b * sA };
  const _Float16* pB[2] = { Btr + (size_t)b * sB, Bti + (size_t)b * sB };

  // ---- persistent per-thread staging descriptors (8 x 16B per K-step) ----
  unsigned long long g8[8];
  uint32_t           l8[8];
#pragma unroll
  for (int t = 0; t < 8; ++t) {
    const int cid = tid + t * 128;                   // 1024 x 16B chunks
    const int p   = cid >> 8;                        // plane 0..3
    const int rem = cid & 255;
    const int r   = rem >> 2;                        // row 0..63
    const int c8  = (rem & 3) * 8;                   // half-offset 0/8/16/24
    const _Float16* g = (p < 2)
        ? pA[p]     + (size_t)(bm * 64 + r) * lda + c8
        : pB[p - 2] + (size_t)(bn * 64 + r) * ldb + c8;
    g8[t] = (unsigned long long)(uintptr_t)g;
    l8[t] = lds_off(&Ls[0][p * LPLN + r * LROW + c8]);
  }
  const uint32_t LBUF = (uint32_t)(4 * LPLN * sizeof(_Float16));

  // Karatsuba accumulators per tile: P=Ar*Br, Q=Ai*Bi, S=(Ar+Ai)*(Br+Bi)
  v8f accP[2][2] = {}, accQ[2][2] = {}, accS[2][2] = {};

  const int nIter = K / 32;

  // prologue: stage tile 0 into buffer 0
#pragma unroll
  for (int t = 0; t < 8; ++t) async_cp16(l8[t], g8[t]);

  int buf = 0;
  for (int i = 0; i < nIter; ++i) {
    if (i > 0) __syncthreads();                      // buffer-reuse guard
    const bool more = (i + 1 < nIter);
    if (more) {                                      // stream tile i+1 now
      const uint32_t lo1 = (uint32_t)(buf ^ 1) * LBUF;
#pragma unroll
      for (int t = 0; t < 8; ++t) {
        g8[t] += 64;                                 // advance 32 halves
        async_cp16(l8[t] + lo1, g8[t]);
      }
      __builtin_prefetch((const void*)(uintptr_t)(g8[0] + 64), 0, 1);
    }
    wait_async(more ? 1 : 0);                        // tile i landed (in-order)
    __syncthreads();

    // ---- fragments: 2x ds_load_b128 per lane, per ISA f16 layouts --------
    const int kbA = (lane < 16) ? 0 : 8;   // A: k {0..7,16..23} / {8..15,24..31}
    const int kbB = (lane < 16) ? 0 : 8;   // B: k {0..7,16..23} / {8..15,24..31}
    v16h aR[2], aI[2], aS[2], bR[2], bI[2], bS[2];
#pragma unroll
    for (int mt = 0; mt < 2; ++mt) {
      const int row = (wm * 32 + mt * 16 + (lane & 15)) * LROW;
      const v8h lr = *(const v8h*)&Ls[buf][0 * LPLN + row + kbA];
      const v8h hr = *(const v8h*)&Ls[buf][0 * LPLN + row + kbA + 16];
      const v8h li = *(const v8h*)&Ls[buf][1 * LPLN + row + kbA];
      const v8h hi = *(const v8h*)&Ls[buf][1 * LPLN + row + kbA + 16];
      aR[mt] = __builtin_shufflevector(lr, hr, 0,1,2,3,4,5,6,7,8,9,10,11,12,13,14,15);
      aI[mt] = __builtin_shufflevector(li, hi, 0,1,2,3,4,5,6,7,8,9,10,11,12,13,14,15);
      aS[mt] = aR[mt] + aI[mt];                      // v_pk_add_f16, co-executes
    }
#pragma unroll
    for (int nt = 0; nt < 2; ++nt) {
      const int row = (wn * 32 + nt * 16 + (lane & 15)) * LROW;
      const v8h lr = *(const v8h*)&Ls[buf][2 * LPLN + row + kbB];
      const v8h hr = *(const v8h*)&Ls[buf][2 * LPLN + row + kbB + 16];
      const v8h li = *(const v8h*)&Ls[buf][3 * LPLN + row + kbB];
      const v8h hi = *(const v8h*)&Ls[buf][3 * LPLN + row + kbB + 16];
      bR[nt] = __builtin_shufflevector(lr, hr, 0,1,2,3,4,5,6,7,8,9,10,11,12,13,14,15);
      bI[nt] = __builtin_shufflevector(li, hi, 0,1,2,3,4,5,6,7,8,9,10,11,12,13,14,15);
      bS[nt] = bR[nt] + bI[nt];
    }

    // ---- 12 WMMAs per wave per K-step (3 per complex tile) ---------------
#pragma unroll
    for (int mt = 0; mt < 2; ++mt)
#pragma unroll
      for (int nt = 0; nt < 2; ++nt) {
        accP[mt][nt] = WMMA_F16(aR[mt], bR[nt], accP[mt][nt]);
        accQ[mt][nt] = WMMA_F16(aI[mt], bI[nt], accQ[mt][nt]);
        accS[mt][nt] = WMMA_F16(aS[mt], bS[nt], accS[mt][nt]);
      }
    buf ^= 1;
  }

  // ---- combine + store (f32 C/D layout: VGPR v -> row v + (lane<16?0:8)) -
#pragma unroll
  for (int mt = 0; mt < 2; ++mt)
#pragma unroll
    for (int nt = 0; nt < 2; ++nt) {
      const v8f cr = accP[mt][nt] - accQ[mt][nt];
      const v8f ci = accS[mt][nt] - accP[mt][nt] - accQ[mt][nt];
      const int col = bn * 64 + wn * 32 + nt * 16 + (lane & 15);
      const int r0  = bm * 64 + wm * 32 + mt * 16 + ((lane < 16) ? 0 : 8);
#pragma unroll
      for (int v = 0; v < 8; ++v) {
        const size_t idx = (size_t)b * sC + (size_t)(r0 + v) * ldc + col;
        if (C16r) { C16r[idx] = (_Float16)cr[v];
                    C16i[idx] = (_Float16)ci[v]; }
        if (C32r) { C32r[idx] = cr[v];
                    C32i[idx] = ci[v]; }
      }
    }
}

// ---------------------------------------------------------------------------
// Setup kernels
// ---------------------------------------------------------------------------

// Forward DFT Ff (384x384) and cropped inverse Gc (128x384), f16 planes.
// Exact twiddle phases via integer (u*r) mod 384.
__global__ void k_build_dft(_Float16* Ffr, _Float16* Ffi,
                            _Float16* Gcr, _Float16* Gci) {
  const int id = blockIdx.x * blockDim.x + threadIdx.x;
  if (id < LDIM * LDIM) {
    const int u = id / LDIM, r = id % LDIM;
    const int t = (u * r) % LDIM;
    const float a = -(float)(TWO_PI) * (float)t / (float)LDIM;
    float s, c; __sincosf(a, &s, &c);
    Ffr[id] = (_Float16)c; Ffi[id] = (_Float16)s;
  } else if (id < LDIM * LDIM + NPX * LDIM) {
    const int e = id - LDIM * LDIM;
    const int m = e / LDIM, u = e % LDIM;
    const int t = ((m + NPX) * u) % LDIM;        // rows 128..255 (crop)
    const float a = (float)(TWO_PI) * (float)t / (float)LDIM;
    float s, c; __sincosf(a, &s, &c);
    const float sc = 1.0f / (float)LDIM;
    Gcr[e] = (_Float16)(c * sc); Gci[e] = (_Float16)(s * sc);
  }
}

// Rayleigh-Sommerfeld kernel (symmetric -> flip is identity, W^T == W),
// zero-padded to 384x384 f16. Phase r/lambda ~3.5e4 cycles -> f64 reduction.
__global__ void k_build_w(_Float16* Wr, _Float16* Wi) {
  const int id = blockIdx.x * blockDim.x + threadIdx.x;
  if (id >= LDIM * LDIM) return;
  const int i = id / LDIM, j = id % LDIM;
  float re = 0.f, im = 0.f;
  if (i < 2 * NPX + 1 && j < 2 * NPX + 1) {
    const double x = ((double)i - NPX) * PIXSZ;
    const double y = ((double)j - NPX) * PIXSZ;
    const double z = DISTZ;
    const double r2 = x * x + y * y + z * z;
    const double r  = sqrt(r2);
    const double f  = (PIXSZ * PIXSZ) * z / r2;
    const double ar = 1.0 / (2.0 * PI_D * r);
    const double bi = -1.0 / WLEN;               // 1/(i*WL) = -i/WL
    const double cyc = r / WLEN;
    const double ph  = (cyc - floor(cyc)) * TWO_PI;
    const double sp = sin(ph), cp = cos(ph);
    re = (float)(f * (ar * cp - bi * sp));
    im = (float)(f * (ar * sp + bi * cp));
  }
  Wr[id] = (_Float16)re; Wi[id] = (_Float16)im;
}

// Y(f16) *= H(f32), write transposed f16 Yt for the next GEMM's B^T operand.
__global__ void k_apply_H(const _Float16* __restrict__ Yr,
                          const _Float16* __restrict__ Yi,
                          const float* __restrict__ Hr,
                          const float* __restrict__ Hi,
                          _Float16* __restrict__ Ytr,
                          _Float16* __restrict__ Yti) {
  const size_t id = (size_t)blockIdx.x * blockDim.x + threadIdx.x;
  const size_t L2 = (size_t)LDIM * LDIM;
  if (id >= (size_t)BATCH * L2) return;
  const size_t e = id % L2, b = id / L2;
  const int u = (int)(e / LDIM), v = (int)(e % LDIM);
  const float yr = (float)Yr[id], yi = (float)Yi[id];
  const float hr = Hr[e], hi = Hi[e];
  const size_t o = b * L2 + (size_t)v * LDIM + u;
  Ytr[o] = (_Float16)(yr * hr - yi * hi);
  Yti[o] = (_Float16)(yr * hi + yi * hr);
}

// Input -> transposed f16 field planes (B^T operand) + snapshot slot 0.
__global__ void k_init(const float* __restrict__ x,
                       _Float16* Xtr, _Float16* Xti, float* out) {
  const size_t id = (size_t)blockIdx.x * blockDim.x + threadIdx.x;
  const size_t NP = (size_t)BATCH * NPX * NPX;
  if (id >= NP) return;
  const size_t b = id / (NPX * NPX);
  const int e = (int)(id % (NPX * NPX));
  const int i = e / NPX, j = e % NPX;
  const float v = x[id];
  const size_t o = b * NPX * NPX + (size_t)j * NPX + i;   // transpose
  Xtr[o] = (_Float16)v; Xti[o] = (_Float16)0.f;
  float* snap = out + NP;
  snap[2 * id] = v; snap[2 * id + 1] = 0.f;
}

// Phase mask (layers 0..4) -> transposed f16 field, snapshot, final |.|^2.
__global__ void k_epilogue(const float* __restrict__ Znr,
                           const float* __restrict__ Zni,
                           const float* __restrict__ phases,
                           _Float16* Xtr, _Float16* Xti,
                           float* out, int layer) {
  const size_t id = (size_t)blockIdx.x * blockDim.x + threadIdx.x;
  const size_t NP = (size_t)BATCH * NPX * NPX;
  if (id >= NP) return;
  const size_t b = id / (NPX * NPX);
  const int e = (int)(id % (NPX * NPX));
  const int i = e / NPX, j = e % NPX;
  float nr = Znr[id], ni = Zni[id];
  if (layer < NLAY) {
    // 2*pi*sigmoid(p); nearest 64->128 resize is p[i>>1][j>>1]
    const float p  = phases[layer * 64 * 64 + (i >> 1) * 64 + (j >> 1)];
    const float cp = (float)(TWO_PI) / (1.0f + __expf(-p));
    float s, c; __sincosf(cp, &s, &c);
    const float zr = nr, zi = ni;
    nr = c * zr - s * zi;
    ni = c * zi + s * zr;
    const size_t o = b * NPX * NPX + (size_t)j * NPX + i;  // transpose
    Xtr[o] = (_Float16)nr; Xti[o] = (_Float16)ni;
  }
  float* snap = out + NP + (size_t)(layer + 1) * 2 * NP;
  snap[2 * id] = nr; snap[2 * id + 1] = ni;
  if (layer == NLAY) out[id] = nr * nr + ni * ni;          // x_abs
}

// ---------------------------------------------------------------------------
// Host launch
// ---------------------------------------------------------------------------
extern "C" void kernel_launch(void* const* d_in, const int* in_sizes, int n_in,
                              void* d_out, int out_size, void* d_ws, size_t ws_size,
                              hipStream_t stream) {
  const float* x      = (const float*)d_in[0];   // [64,128,128]
  const float* phases = (const float*)d_in[1];   // [5,64,64]
  float* out = (float*)d_out;
  (void)ws_size; (void)out_size; (void)in_sizes; (void)n_in;

  const size_t L2 = (size_t)LDIM * LDIM;         // 147456
  const size_t G2 = (size_t)NPX * LDIM;          // 49152
  const size_t XN = (size_t)BATCH * NPX * NPX;   // 1048576
  const size_t TN = (size_t)BATCH * LDIM * NPX;  // 3145728
  const size_t YN = (size_t)BATCH * LDIM * LDIM; // 9437184

  // f16 region
  _Float16* h16 = (_Float16*)d_ws;
  size_t o = 0;
  _Float16 *Ffr = h16 + o; o += L2;  _Float16 *Ffi = h16 + o; o += L2;
  _Float16 *Gcr = h16 + o; o += G2;  _Float16 *Gci = h16 + o; o += G2;
  _Float16 *Wr  = h16 + o; o += L2;  _Float16 *Wi  = h16 + o; o += L2;
  _Float16 *Xtr = h16 + o; o += XN;  _Float16 *Xti = h16 + o; o += XN;
  _Float16 *Tr  = h16 + o; o += TN;  _Float16 *Ti  = h16 + o; o += TN;  // T/Z/Th
  _Float16 *Yr  = h16 + o; o += YN;  _Float16 *Yi  = h16 + o; o += YN;
  _Float16 *Ytr = h16 + o; o += YN;  _Float16 *Yti = h16 + o; o += YN;
  // f32 region (after f16, offset in halves -> even, 4B aligned)
  float* f32 = (float*)(h16 + ((o + 7) & ~(size_t)7));
  size_t o2 = 0;
  float *Hr  = f32 + o2; o2 += L2;   float *Hi  = f32 + o2; o2 += L2;
  float *Xnr = f32 + o2; o2 += XN;   float *Xni = f32 + o2; o2 += XN;

  const int BS = 256;

  // --- setup: DFT matrices, RS kernel, H = Ff * W * Ff^T -----------------
  k_build_dft<<<dim3((unsigned)((L2 + G2 + BS - 1) / BS)), BS, 0, stream>>>(
      Ffr, Ffi, Gcr, Gci);
  k_build_w<<<dim3((unsigned)((L2 + BS - 1) / BS)), BS, 0, stream>>>(Wr, Wi);
  // Th = Ff * W^T (W symmetric) -> T buffer, f16
  k_cgemm<<<dim3(LDIM / 64, LDIM / 64, 1), 128, 0, stream>>>(
      Ffr, Ffi, 0, LDIM,  Wr, Wi, 0, LDIM,
      Tr, Ti, nullptr, nullptr, 0, LDIM, LDIM);
  // H = Th * Ff^T -> f32
  k_cgemm<<<dim3(LDIM / 64, LDIM / 64, 1), 128, 0, stream>>>(
      Tr, Ti, 0, LDIM,  Ffr, Ffi, 0, LDIM,
      nullptr, nullptr, Hr, Hi, 0, LDIM, LDIM);

  // --- initial field + snapshot 0 ---
  k_init<<<dim3((unsigned)((XN + BS - 1) / BS)), BS, 0, stream>>>(x, Xtr, Xti, out);

  // --- 6 propagations ---
  for (int n = 0; n <= NLAY; ++n) {
    // T = Fc * X^T'   (M=384,N=128,K=128), Fc = Ff[:, :128] via lda
    k_cgemm<<<dim3(NPX / 64, LDIM / 64, BATCH), 128, 0, stream>>>(
        Ffr, Ffi, 0, LDIM,
        Xtr, Xti, (long long)(NPX * NPX), NPX,
        Tr, Ti, nullptr, nullptr, (long long)(LDIM * NPX), NPX, NPX);
    // Y = T * Fc^T    (M=384,N=384,K=128)
    k_cgemm<<<dim3(LDIM / 64, LDIM / 64, BATCH), 128, 0, stream>>>(
        Tr, Ti, (long long)(LDIM * NPX), NPX,
        Ffr, Ffi, 0, LDIM,
        Yr, Yi, nullptr, nullptr, (long long)(LDIM * LDIM), LDIM, NPX);
    // Y *= H, transpose -> Yt (f16); Y is L2-resident (38 MB < 192 MB)
    k_apply_H<<<dim3((unsigned)((YN + BS - 1) / BS)), BS, 0, stream>>>(
        Yr, Yi, Hr, Hi, Ytr, Yti);
    // Z = Gc * Yt^T   (M=128,N=384,K=384) -> T buffer
    k_cgemm<<<dim3(LDIM / 64, NPX / 64, BATCH), 128, 0, stream>>>(
        Gcr, Gci, 0, LDIM,
        Ytr, Yti, (long long)(LDIM * LDIM), LDIM,
        Tr, Ti, nullptr, nullptr, (long long)(NPX * LDIM), LDIM, LDIM);
    // Xn = Z * Gc^T   (M=128,N=128,K=384) -> f32
    k_cgemm<<<dim3(NPX / 64, NPX / 64, BATCH), 128, 0, stream>>>(
        Tr, Ti, (long long)(NPX * LDIM), LDIM,
        Gcr, Gci, 0, LDIM,
        nullptr, nullptr, Xnr, Xni, (long long)(NPX * NPX), NPX, LDIM);
    // mask + snapshot (+ |.|^2 on final)
    k_epilogue<<<dim3((unsigned)((XN + BS - 1) / BS)), BS, 0, stream>>>(
        Xnr, Xni, phases, Xtr, Xti, out, n);
  }
}
